// NeuFlow_82712480187036
// MI455X (gfx1250) — matrix operand order repaired and checked by reading, not compiled, not run on hardware
//
#include <hip/hip_runtime.h>
#include <hip/hip_bf16.h>
#include <math.h>

// ---------------------------------------------------------------------------
// CDNA5 (gfx1250) implementation of the NeuFlow-style pipeline.
// All GEMM-like math (feature correlation + every 3x3 conv as implicit GEMM)
// runs on v_wmma_f32_16x16x32_f16. Activations/weights are packed to fp16 in
// pixel-major, K-contiguous, 32-padded layouts so WMMA fragments load as two
// contiguous 16B chunks per lane. vol8 (116MB fp16) stays L2-resident.
// Conv halo taps use a zeroed dummy row => uniform control flow (no exec
// masking around WMMA). Conv epilogues write the packed fp16 layout directly,
// removing fp32 round-trips between chained convs.
// ---------------------------------------------------------------------------

typedef __attribute__((ext_vector_type(16))) _Float16 v16h;
typedef __attribute__((ext_vector_type(8)))  float    v8f;
typedef __attribute__((ext_vector_type(4)))  float    f4;

union V16H { v16h v; f4 f[2]; };

__device__ __forceinline__ float gelu_tanh(float x) {
    const float k0 = 0.7978845608028654f, k1 = 0.044715f;
    return 0.5f * x * (1.0f + tanhf(k0 * (x + k1 * x * x * x)));
}

// ---------------------------------------------------------------------------
// Packing kernels
// ---------------------------------------------------------------------------

// NCHW fp32 -> (B*P, Cpad) fp16 pixel-major rows (zero padded channels).
__global__ void pack_act(const float* __restrict__ src, _Float16* __restrict__ dst,
                         int Bb, int C, int P, int Cpad, int relu)
{
    size_t idx = (size_t)blockIdx.x * blockDim.x + threadIdx.x;
    size_t total = (size_t)Bb * P * Cpad;
    if (idx >= total) return;
    int c = (int)(idx % Cpad);
    size_t bp = idx / Cpad;
    int b = (int)(bp / P), p = (int)(bp % P);
    float v = 0.0f;
    if (c < C) {
        v = src[((size_t)b * C + c) * P + p];
        if (relu) v = fmaxf(v, 0.0f);
    }
    dst[idx] = (_Float16)v;
}

// (Co,Ci,3,3) fp32 -> (Co16, 9*Cpad) fp16, K ordered (tap, cin), zero padded.
__global__ void pack_weights(const float* __restrict__ w, _Float16* __restrict__ wp,
                             int Co, int Ci, int Cpad, int Co16)
{
    int idx = blockIdx.x * blockDim.x + threadIdx.x;
    int total = Co16 * 9 * Cpad;
    if (idx >= total) return;
    int c = idx % Cpad;
    int t = (idx / Cpad) % 9;
    int o = idx / (9 * Cpad);
    float v = 0.0f;
    if (o < Co && c < Ci) v = w[((size_t)o * Ci + c) * 9 + t];
    wp[idx] = (_Float16)v;
}

// ---------------------------------------------------------------------------
// Correlation GEMM:  out[b][p][q] = scale * sum_c Ah[b][p][c] * Bh[b][q][c]
// 4 waves per block; wave w computes the 16x16 tile at (pbase, qbase + 16w).
// ---------------------------------------------------------------------------
template <typename OutT>
__global__ void __launch_bounds__(128) corr_gemm_wmma(
    const _Float16* __restrict__ Ah, const _Float16* __restrict__ Bh,
    OutT* __restrict__ out, int P, int C, float scale)
{
    const int lane = threadIdx.x & 31;
    const int wave = threadIdx.x >> 5;
    const int col = lane & 15, hi = lane >> 4;
    const int pbase = blockIdx.x * 16;
    const int qbase = (blockIdx.y * 4 + wave) * 16;
    const int b = blockIdx.z;

    const _Float16* arow = Ah + ((size_t)b * P + pbase + col) * C;
    const _Float16* brow = Bh + ((size_t)b * P + qbase + col) * C;

    v8f acc = {};
    for (int k = 0; k < C; k += 32) {
        V16H a, bf;
        { // A frag: lanes 0-15 K={k..k+7, k+16..k+23}; lanes 16-31 shifted by 8
            const _Float16* r = arow + k + hi * 8;
            a.f[0] = *(const f4*)(r);
            a.f[1] = *(const f4*)(r + 16);
        }
        { // B frag: lanes 0-15 K={k..k+7, k+16..k+23}; lanes 16-31 shifted by 8
            const _Float16* r = brow + k + hi * 8;
            bf.f[0] = *(const f4*)(r);
            bf.f[1] = *(const f4*)(r + 16);
        }
        acc = __builtin_amdgcn_wmma_f32_16x16x32_f16(
            false, a.v, false, bf.v, (short)0, acc, false, false);
    }

    OutT* o = out + ((size_t)b * P + pbase) * P + qbase;
    for (int j = 0; j < 8; ++j) {
        const int row = hi * 8 + j;
        o[(size_t)row * P + col] = (OutT)(acc[j] * scale);
    }
}

// ---------------------------------------------------------------------------
// 3x3 SAME conv as implicit GEMM.  M=out-channels, N=16 pixels, K=9*Cpad.
// 4 waves per block cover 4 adjacent pixel tiles. Halo taps read a zeroed
// dummy row (uniform control flow -> no exec masking around WMMA).
// Epilogue modes: 0=none 1=gelu 2=tanh 3=x*g+b. Writes fp32 NCHW (Y) and/or
// packed fp16 pixel-major (Yp, CoPad stride) outputs.
// ---------------------------------------------------------------------------
__global__ void __launch_bounds__(128) conv3x3_wmma(
    const _Float16* __restrict__ Xp,   // (Bb*P, Cpad) pixel-major activations
    const _Float16* __restrict__ Wp,   // (Co16, 9*Cpad) packed weights
    const _Float16* __restrict__ zrow, // zeroed row, >= Cpad halves
    float* __restrict__ Y,             // optional (Bb, Co, P) fp32 output
    _Float16* __restrict__ Yp,         // optional (Bb*P, CoPad) fp16 output
    const float* __restrict__ gamma, const float* __restrict__ beta,
    int H, int W, int Cpad, int Co, int CoPad, int mode)
{
    const int lane = threadIdx.x & 31;
    const int wave = threadIdx.x >> 5;
    const int col = lane & 15, hi = lane >> 4;
    const int pbase = (blockIdx.x * 4 + wave) * 16;
    const int obase = blockIdx.y * 16;
    const int bb = blockIdx.z;
    const int P = H * W;

    const int p  = pbase + col;          // this lane's pixel (N index)
    const int py = p / W, px = p % W;
    const int ldw = 9 * Cpad;
    const _Float16* wrow = Wp + (size_t)(obase + col) * ldw;

    v8f acc = {};
    for (int t = 0; t < 9; ++t) {
        const int dy = t / 3 - 1, dx = t % 3 - 1;
        const int ny = py + dy, nx = px + dx;
        const bool valid = (ny >= 0) && (ny < H) && (nx >= 0) && (nx < W);
        const _Float16* xrow = valid
            ? (Xp + ((size_t)bb * P + (size_t)ny * W + nx) * Cpad)
            : zrow;
        __builtin_prefetch((const void*)xrow, 0, 0);  // global_prefetch_b8

        for (int c = 0; c < Cpad; c += 32) {
            V16H a, bf;
            { // weights fragment
                const _Float16* r = wrow + t * Cpad + c + hi * 8;
                a.f[0] = *(const f4*)(r);
                a.f[1] = *(const f4*)(r + 16);
            }
            { // im2col fragment (zrow for halo => unconditional loads)
                const _Float16* r = xrow + c + hi * 8;
                bf.f[0] = *(const f4*)(r);
                bf.f[1] = *(const f4*)(r + 16);
            }
            acc = __builtin_amdgcn_wmma_f32_16x16x32_f16(
                false, a.v, false, bf.v, (short)0, acc, false, false);
        }
    }

    for (int j = 0; j < 8; ++j) {
        const int oc = obase + hi * 8 + j;
        if (oc < Co) {
            float v = acc[j];
            if (mode == 1)      v = gelu_tanh(v);
            else if (mode == 2) v = tanhf(v);
            else if (mode == 3) v = v * gamma[oc] + beta[oc];
            if (Y)  Y[((size_t)bb * Co + oc) * P + p] = v;
            if (Yp) Yp[((size_t)bb * P + p) * CoPad + oc] = (_Float16)v;
        }
    }
}

// ---------------------------------------------------------------------------
// Softmax over correlation rows + expected-coordinate flow (global_corr_flow).
// One block (256 thr) per (p, b) row of length P.
// ---------------------------------------------------------------------------
__global__ void softmax_flow_kernel(const float* __restrict__ corr,
                                    float* __restrict__ flow,
                                    int P, int H, int W)
{
    const int p = blockIdx.x;
    const int b = blockIdx.y;
    const int tid = threadIdx.x;
    const float* row = corr + ((size_t)b * P + p) * P;
    __shared__ float red[256];

    float mx = -3.4e38f;
    for (int q = tid; q < P; q += 256) mx = fmaxf(mx, row[q]);
    red[tid] = mx; __syncthreads();
    for (int s = 128; s > 0; s >>= 1) { if (tid < s) red[tid] = fmaxf(red[tid], red[tid + s]); __syncthreads(); }
    mx = red[0]; __syncthreads();

    float sum = 0.0f, sx = 0.0f, sy = 0.0f;
    for (int q = tid; q < P; q += 256) {
        float e = __expf(row[q] - mx);
        sum += e;
        sx  += e * (float)(q % W);
        sy  += e * (float)(q / W);
    }
    red[tid] = sum; __syncthreads();
    for (int s = 128; s > 0; s >>= 1) { if (tid < s) red[tid] += red[tid + s]; __syncthreads(); }
    sum = red[0]; __syncthreads();
    red[tid] = sx; __syncthreads();
    for (int s = 128; s > 0; s >>= 1) { if (tid < s) red[tid] += red[tid + s]; __syncthreads(); }
    sx = red[0]; __syncthreads();
    red[tid] = sy; __syncthreads();
    for (int s = 128; s > 0; s >>= 1) { if (tid < s) red[tid] += red[tid + s]; __syncthreads(); }
    sy = red[0];

    if (tid == 0) {
        float inv = 1.0f / sum;
        flow[((size_t)b * 2 + 0) * P + p] = sx * inv - (float)(p % W);
        flow[((size_t)b * 2 + 1) * P + p] = sy * inv - (float)(p / W);
    }
}

// ---------------------------------------------------------------------------
// corr_lookup (81-tap bilinear) fused with refine-input assembly:
// X[p] = [corrs(81) | flow(2) | ctx(64) | ictx(64) | zero-pad] packed fp16.
// ---------------------------------------------------------------------------
template <typename VolT>
__global__ void build_refine_input(
    const VolT* __restrict__ vol, const float* __restrict__ flow,
    const float* __restrict__ ctx, const float* __restrict__ ictx,
    _Float16* __restrict__ Xp, int BP, int H, int W, int Cpad, int relu_ctx)
{
    int idx = blockIdx.x * blockDim.x + threadIdx.x;
    if (idx >= BP) return;
    const int P = H * W;
    const int b = idx / P, p = idx % P;
    const float fx = flow[((size_t)b * 2 + 0) * P + p];
    const float fy = flow[((size_t)b * 2 + 1) * P + p];
    const float cx = (float)(p % W) + fx;
    const float cy = (float)(p / W) + fy;
    const VolT* vrow = vol + ((size_t)b * P + p) * P;
    _Float16* xo = Xp + (size_t)idx * Cpad;

    auto g = [&](int xi, int yi) -> float {
        if (xi < 0 || xi >= W || yi < 0 || yi >= H) return 0.0f;
        return (float)vrow[yi * W + xi];
    };

    for (int t = 0; t < 81; ++t) {
        const float x = cx + (float)(t % 9 - 4);
        const float y = cy + (float)(t / 9 - 4);
        const float x0f = floorf(x), y0f = floorf(y);
        const float wx = x - x0f, wy = y - y0f;
        const int xi = (int)x0f, yi = (int)y0f;
        const float v00 = g(xi, yi),     v01 = g(xi + 1, yi);
        const float v10 = g(xi, yi + 1), v11 = g(xi + 1, yi + 1);
        const float v = (v00 * (1.0f - wx) + v01 * wx) * (1.0f - wy)
                      + (v10 * (1.0f - wx) + v11 * wx) * wy;
        xo[t] = (_Float16)v;
    }
    xo[81] = (_Float16)fx;
    xo[82] = (_Float16)fy;
    for (int c = 0; c < 64; ++c) {
        float v = ctx[((size_t)b * 64 + c) * P + p];
        if (relu_ctx) v = fmaxf(v, 0.0f);
        xo[83 + c] = (_Float16)v;
    }
    for (int c = 0; c < 64; ++c)
        xo[147 + c] = (_Float16)ictx[((size_t)b * 64 + c) * P + p];
    for (int c = 211; c < Cpad; ++c) xo[c] = (_Float16)0.0f;
}

// ---------------------------------------------------------------------------
// Merge-input assembly: channels [feats8(128) | up2(feats16)(128)] -> fp16.
// Batches 0..B-1 use f0, B..2B-1 use f1 (matches jnp.concatenate on batch).
// ---------------------------------------------------------------------------
__global__ void assemble_merge_input(
    const float* __restrict__ f0s8, const float* __restrict__ f1s8,
    const float* __restrict__ f0s16, const float* __restrict__ f1s16,
    _Float16* __restrict__ Xp, int B, int H8, int W8, int H16, int W16)
{
    const int P8 = H8 * W8, P16 = H16 * W16;
    size_t idx = (size_t)blockIdx.x * blockDim.x + threadIdx.x;
    size_t total = (size_t)(2 * B) * P8 * 256;
    if (idx >= total) return;
    int c = (int)(idx % 256);
    size_t bp = idx / 256;
    int bb = (int)(bp / P8), p = (int)(bp % P8);
    int b = bb % B;
    const float* fs8  = (bb < B) ? f0s8  : f1s8;
    const float* fs16 = (bb < B) ? f0s16 : f1s16;
    float v;
    if (c < 128) {
        v = fs8[((size_t)b * 128 + c) * P8 + p];
    } else {
        int y16 = (p / W8) >> 1, x16 = (p % W8) >> 1;
        v = fs16[((size_t)b * 128 + (c - 128)) * P16 + (size_t)y16 * W16 + x16];
    }
    Xp[idx] = (_Float16)v;
}

// Context-merge input: [relu(ctx8)(64) | up2(relu(ctx16))(64)] -> fp16.
__global__ void assemble_cmerge_input(
    const float* __restrict__ ctx8, const float* __restrict__ ctx16,
    _Float16* __restrict__ Xp, int B, int H8, int W8, int H16, int W16)
{
    const int P8 = H8 * W8, P16 = H16 * W16;
    size_t idx = (size_t)blockIdx.x * blockDim.x + threadIdx.x;
    size_t total = (size_t)B * P8 * 128;
    if (idx >= total) return;
    int c = (int)(idx % 128);
    size_t bp = idx / 128;
    int b = (int)(bp / P8), p = (int)(bp % P8);
    float v;
    if (c < 64) {
        v = ctx8[((size_t)b * 64 + c) * P8 + p];
    } else {
        int y16 = (p / W8) >> 1, x16 = (p % W8) >> 1;
        v = ctx16[((size_t)b * 64 + (c - 64)) * P16 + (size_t)y16 * W16 + x16];
    }
    Xp[idx] = (_Float16)fmaxf(v, 0.0f);
}

// flow8 = 2 * up2(flow16)
__global__ void upflow2_kernel(const float* __restrict__ flow16,
                               float* __restrict__ flow8, int B, int H16, int W16)
{
    const int H8 = H16 * 2, W8 = W16 * 2;
    const int P8 = H8 * W8, P16 = H16 * W16;
    size_t idx = (size_t)blockIdx.x * blockDim.x + threadIdx.x;
    size_t total = (size_t)B * 2 * P8;
    if (idx >= total) return;
    int p = (int)(idx % P8);
    int c = (int)((idx / P8) % 2);
    int b = (int)(idx / ((size_t)2 * P8));
    int y8 = p / W8, x8 = p % W8;
    flow8[idx] = 2.0f * flow16[((size_t)b * 2 + c) * P16 + (size_t)(y8 >> 1) * W16 + (x8 >> 1)];
}

__global__ void add_inplace(float* __restrict__ a, const float* __restrict__ d, int n)
{
    int i = blockIdx.x * blockDim.x + threadIdx.x;
    if (i < n) a[i] += d[i];
}

// half-pixel bilinear x8 upsample, then *8 (jax.image.resize semantics)
__global__ void resize_bilinear8(const float* __restrict__ src, float* __restrict__ dst,
                                 int B, int Hs, int Ws)
{
    const int Hd = Hs * 8, Wd = Ws * 8;
    size_t idx = (size_t)blockIdx.x * blockDim.x + threadIdx.x;
    size_t total = (size_t)B * 2 * Hd * Wd;
    if (idx >= total) return;
    int xo = (int)(idx % Wd);
    int yo = (int)((idx / Wd) % Hd);
    int c  = (int)((idx / ((size_t)Wd * Hd)) % 2);
    int b  = (int)(idx / ((size_t)2 * Wd * Hd));
    float sx = (xo + 0.5f) * 0.125f - 0.5f;
    float sy = (yo + 0.5f) * 0.125f - 0.5f;
    float x0f = floorf(sx), y0f = floorf(sy);
    float wx = sx - x0f, wy = sy - y0f;
    int x0 = (int)x0f, y0 = (int)y0f;
    int x0c = min(max(x0, 0), Ws - 1), x1c = min(max(x0 + 1, 0), Ws - 1);
    int y0c = min(max(y0, 0), Hs - 1), y1c = min(max(y0 + 1, 0), Hs - 1);
    const float* s = src + ((size_t)b * 2 + c) * Hs * Ws;
    float v = (s[(size_t)y0c * Ws + x0c] * (1.0f - wx) + s[(size_t)y0c * Ws + x1c] * wx) * (1.0f - wy)
            + (s[(size_t)y1c * Ws + x0c] * (1.0f - wx) + s[(size_t)y1c * Ws + x1c] * wx) * wy;
    dst[idx] = v * 8.0f;
}

// ---------------------------------------------------------------------------
// Host-side orchestration
// ---------------------------------------------------------------------------
extern "C" void kernel_launch(void* const* d_in, const int* in_sizes, int n_in,
                              void* d_out, int out_size, void* d_ws, size_t ws_size,
                              hipStream_t stream)
{
    (void)in_sizes; (void)n_in; (void)out_size; (void)ws_size;

    // Problem constants (from setup_inputs).
    const int B = 2;
    const int H16 = 28, W16 = 48, P16 = H16 * W16;      // 1344
    const int H8 = 56,  W8 = 96,  P8  = H8 * W8;        // 5376
    const int C = 128;
    const int INPAD = 224;                               // 211 -> pad 224
    const int ITERS16 = 1, ITERS8 = 8;                   // fixed by setup_inputs
    const float invsqc = 0.08838834764831845f;           // 1/sqrt(128)

    const float* f0s16 = (const float*)d_in[0];
    const float* f1s16 = (const float*)d_in[1];
    const float* ctxs16 = (const float*)d_in[2];
    const float* f0s8  = (const float*)d_in[3];
    const float* f1s8  = (const float*)d_in[4];
    const float* ctxs8 = (const float*)d_in[5];
    const float* merge_w1 = (const float*)d_in[6];
    const float* merge_w2 = (const float*)d_in[7];
    const float* merge_g  = (const float*)d_in[8];
    const float* merge_b  = (const float*)d_in[9];
    const float* cmerge_w1 = (const float*)d_in[10];
    const float* cmerge_w2 = (const float*)d_in[11];
    const float* cmerge_g  = (const float*)d_in[12];
    const float* cmerge_b  = (const float*)d_in[13];
    const float* r16_w_in   = (const float*)d_in[14];
    const float* r16_w_hid  = (const float*)d_in[15];
    const float* r16_w_ctx  = (const float*)d_in[16];
    const float* r16_w_flow = (const float*)d_in[17];
    const float* r8_w_in    = (const float*)d_in[18];
    const float* r8_w_hid   = (const float*)d_in[19];
    const float* r8_w_ctx   = (const float*)d_in[20];
    const float* r8_w_flow  = (const float*)d_in[21];

    // Workspace bump allocator (256B aligned).
    char* ws = (char*)d_ws;
    size_t off = 0;
    auto alloc = [&](size_t bytes) -> void* {
        void* p = ws + off;
        off = (off + bytes + 255) & ~(size_t)255;
        return p;
    };

    _Float16* zrow  = (_Float16*)alloc(256 * 2);        // zeroed halo row
    _Float16* f0h16 = (_Float16*)alloc((size_t)B * P16 * 128 * 2);
    _Float16* f1h16 = (_Float16*)alloc((size_t)B * P16 * 128 * 2);
    float*    corr16 = (float*)alloc((size_t)B * P16 * P16 * 4);
    float*    flow16 = (float*)alloc((size_t)B * 2 * P16 * 4);
    float*    flow8  = (float*)alloc((size_t)B * 2 * P8 * 4);
    float*    ictx16 = (float*)alloc((size_t)B * 64 * P16 * 4);
    float*    ictx8  = (float*)alloc((size_t)B * 64 * P8 * 4);
    _Float16* X16p  = (_Float16*)alloc((size_t)B * P16 * INPAD * 2);
    _Float16* h1p16 = (_Float16*)alloc((size_t)B * P16 * 128 * 2);
    _Float16* h2p16 = (_Float16*)alloc((size_t)B * P16 * 128 * 2);
    float*    dflow16 = (float*)alloc((size_t)B * 2 * P16 * 4);
    _Float16* Xm    = (_Float16*)alloc((size_t)2 * B * P8 * 256 * 2);
    _Float16* mhp   = (_Float16*)alloc((size_t)2 * B * P8 * 128 * 2);
    _Float16* mp    = (_Float16*)alloc((size_t)2 * B * P8 * 128 * 2);  // merged, packed
    _Float16* vol8  = (_Float16*)alloc((size_t)B * P8 * P8 * 2);       // 116 MB, L2-resident
    _Float16* Xc    = (_Float16*)alloc((size_t)B * P8 * 128 * 2);
    _Float16* cmhp  = (_Float16*)alloc((size_t)B * P8 * 64 * 2);
    float*    ctx8m = (float*)alloc((size_t)B * 64 * P8 * 4);
    _Float16* X8p   = (_Float16*)alloc((size_t)B * P8 * INPAD * 2);
    _Float16* h1p8  = (_Float16*)alloc((size_t)B * P8 * 96 * 2);
    _Float16* h2p8  = (_Float16*)alloc((size_t)B * P8 * 96 * 2);
    float*    dflow8 = (float*)alloc((size_t)B * 2 * P8 * 4);

    // merged[:B] / merged[B:] are contiguous slices of the packed buffer.
    _Float16* m0h = mp;
    _Float16* m1h = mp + (size_t)B * P8 * 128;

    // Packed weights.
    auto alloc_w = [&](int Co, int Cpad) -> _Float16* {
        int co16 = (Co + 15) & ~15;
        return (_Float16*)alloc((size_t)co16 * 9 * Cpad * 2);
    };
    _Float16* wp_m1 = alloc_w(128, 256);
    _Float16* wp_m2 = alloc_w(128, 128);
    _Float16* wp_c1 = alloc_w(64, 128);
    _Float16* wp_c2 = alloc_w(64, 64);
    _Float16* wp_r16_in   = alloc_w(128, INPAD);
    _Float16* wp_r16_hid  = alloc_w(128, 128);
    _Float16* wp_r16_ctx  = alloc_w(64, 128);
    _Float16* wp_r16_flow = alloc_w(2, 128);
    _Float16* wp_r8_in    = alloc_w(96, INPAD);
    _Float16* wp_r8_hid   = alloc_w(96, 96);
    _Float16* wp_r8_ctx   = alloc_w(64, 96);
    _Float16* wp_r8_flow  = alloc_w(2, 96);

    auto pack_w = [&](const float* src, _Float16* dst, int Co, int Ci, int Cpad) {
        int co16 = (Co + 15) & ~15;
        int n = co16 * 9 * Cpad;
        pack_weights<<<(n + 255) / 256, 256, 0, stream>>>(src, dst, Co, Ci, Cpad, co16);
    };
    auto pack_a = [&](const float* src, _Float16* dst, int Bb, int Cc, int P, int Cpad, int relu) {
        size_t n = (size_t)Bb * P * Cpad;
        pack_act<<<(unsigned)((n + 255) / 256), 256, 0, stream>>>(src, dst, Bb, Cc, P, Cpad, relu);
    };
    // 4 pixel-tiles (64 px) per block in x.
    auto conv = [&](const _Float16* Xp, const _Float16* Wp,
                    float* Y, _Float16* Yp,
                    int Bb, int H, int W, int Cpad, int Co, int CoPad, int mode,
                    const float* g, const float* bt) {
        dim3 grid((H * W) / 64, (Co + 15) / 16, Bb);
        conv3x3_wmma<<<grid, 128, 0, stream>>>(Xp, Wp, zrow, Y, Yp, g, bt,
                                               H, W, Cpad, Co, CoPad, mode);
    };

    // ---- weight packing + zero inits (once per call) ----
    hipMemsetAsync(zrow, 0, 256 * 2, stream);
    pack_w(merge_w1, wp_m1, 128, 256, 256);
    pack_w(merge_w2, wp_m2, 128, 128, 128);
    pack_w(cmerge_w1, wp_c1, 64, 128, 128);
    pack_w(cmerge_w2, wp_c2, 64, 64, 64);
    pack_w(r16_w_in,   wp_r16_in,   128, 211, INPAD);
    pack_w(r16_w_hid,  wp_r16_hid,  128, 128, 128);
    pack_w(r16_w_ctx,  wp_r16_ctx,  64, 128, 128);
    pack_w(r16_w_flow, wp_r16_flow, 2, 128, 128);
    pack_w(r8_w_in,    wp_r8_in,    96, 211, INPAD);
    pack_w(r8_w_hid,   wp_r8_hid,   96, 96, 96);
    pack_w(r8_w_ctx,   wp_r8_ctx,   64, 96, 96);
    pack_w(r8_w_flow,  wp_r8_flow,  2, 96, 96);

    hipMemsetAsync(ictx16, 0, (size_t)B * 64 * P16 * 4, stream);
    hipMemsetAsync(ictx8,  0, (size_t)B * 64 * P8  * 4, stream);

    // ---- s16 stage: global correlation + softmax flow ----
    pack_a(f0s16, f0h16, B, 128, P16, 128, 0);
    pack_a(f1s16, f1h16, B, 128, P16, 128, 0);
    corr_gemm_wmma<float><<<dim3(P16 / 16, P16 / 64, B), 128, 0, stream>>>(
        f0h16, f1h16, corr16, P16, C, invsqc);
    softmax_flow_kernel<<<dim3(P16, B), 256, 0, stream>>>(corr16, flow16, P16, H16, W16);

    // ---- s16 refine iterations ----
    for (int it = 0; it < ITERS16; ++it) {
        int bp = B * P16;
        build_refine_input<float><<<(bp + 127) / 128, 128, 0, stream>>>(
            corr16, flow16, ctxs16, ictx16, X16p, bp, H16, W16, INPAD, /*relu_ctx=*/1);
        conv(X16p,  wp_r16_in,  nullptr, h1p16, B, H16, W16, INPAD, 128, 128, 1, nullptr, nullptr);
        conv(h1p16, wp_r16_hid, nullptr, h2p16, B, H16, W16, 128, 128, 128, 1, nullptr, nullptr);
        conv(h2p16, wp_r16_ctx,  ictx16,  nullptr, B, H16, W16, 128, 64, 0, 2, nullptr, nullptr);
        conv(h2p16, wp_r16_flow, dflow16, nullptr, B, H16, W16, 128, 2, 0, 0, nullptr, nullptr);
        add_inplace<<<(B * 2 * P16 + 255) / 256, 256, 0, stream>>>(flow16, dflow16, B * 2 * P16);
    }

    // ---- upsample flow, merge features & context at s8 ----
    {
        size_t n = (size_t)B * 2 * P8;
        upflow2_kernel<<<(unsigned)((n + 255) / 256), 256, 0, stream>>>(flow16, flow8, B, H16, W16);
    }
    {
        size_t n = (size_t)2 * B * P8 * 256;
        assemble_merge_input<<<(unsigned)((n + 255) / 256), 256, 0, stream>>>(
            f0s8, f1s8, f0s16, f1s16, Xm, B, H8, W8, H16, W16);
    }
    conv(Xm,  wp_m1, nullptr, mhp, 2 * B, H8, W8, 256, 128, 128, 1, nullptr, nullptr);
    conv(mhp, wp_m2, nullptr, mp,  2 * B, H8, W8, 128, 128, 128, 3, merge_g, merge_b);

    corr_gemm_wmma<_Float16><<<dim3(P8 / 16, P8 / 64, B), 128, 0, stream>>>(
        m0h, m1h, vol8, P8, C, invsqc);

    {
        size_t n = (size_t)B * P8 * 128;
        assemble_cmerge_input<<<(unsigned)((n + 255) / 256), 256, 0, stream>>>(
            ctxs8, ctxs16, Xc, B, H8, W8, H16, W16);
    }
    conv(Xc,   wp_c1, nullptr, cmhp, B, H8, W8, 128, 64, 64, 1, nullptr, nullptr);
    conv(cmhp, wp_c2, ctx8m, nullptr, B, H8, W8, 64, 64, 0, 3, cmerge_g, cmerge_b);

    // ---- s8 refine iterations ----
    for (int it = 0; it < ITERS8; ++it) {
        int bp = B * P8;
        build_refine_input<_Float16><<<(bp + 127) / 128, 128, 0, stream>>>(
            vol8, flow8, ctx8m, ictx8, X8p, bp, H8, W8, INPAD, /*relu_ctx=*/0);
        conv(X8p,  wp_r8_in,  nullptr, h1p8, B, H8, W8, INPAD, 96, 96, 1, nullptr, nullptr);
        conv(h1p8, wp_r8_hid, nullptr, h2p8, B, H8, W8, 96, 96, 96, 1, nullptr, nullptr);
        conv(h2p8, wp_r8_ctx,  ictx8,  nullptr, B, H8, W8, 96, 64, 0, 2, nullptr, nullptr);
        conv(h2p8, wp_r8_flow, dflow8, nullptr, B, H8, W8, 96, 2, 0, 0, nullptr, nullptr);
        add_inplace<<<(B * 2 * P8 + 255) / 256, 256, 0, stream>>>(flow8, dflow8, B * 2 * P8);
    }

    // ---- final x8 bilinear upsample (+ *8) into d_out ----
    {
        size_t n = (size_t)B * 2 * (H8 * 8) * (W8 * 8);
        resize_bilinear8<<<(unsigned)((n + 255) / 256), 256, 0, stream>>>(
            flow8, (float*)d_out, B, H8, W8);
    }
}
